// SigGNN_31250182046220
// MI455X (gfx1250) — compile-verified, matrix operand order checked by reading, not run on hardware
//
#include <hip/hip_runtime.h>
#include <hip/hip_bf16.h>
#include <math.h>

// ---------------- problem dims ----------------
#define NREAL  30490
#define NPAD   30496          // 16*1906
#define TDIM   90
#define CIN    2
#define EEDGE  487840
#define HID    128
#define NH     4
#define DHD    32
#define KIN    320            // 292 (252 sig + 40 emb) padded to 32-multiple
#define HOR    28
#define HORP   32
#define MLPH   256
#define LAYERS 3
#define ETYPES 3

typedef __attribute__((ext_vector_type(16))) __bf16 v16bf;
typedef __attribute__((ext_vector_type(8)))  float  v8f;

union Frag16 {                 // 8 VGPRs <-> two 128-bit loads
  v16bf v;
  uint4 q[2];
};

// ---------------- helpers ----------------
__device__ __forceinline__ float gelu_f(float x) {
  return 0.5f * x * (1.0f + erff(x * 0.70710678118654752f));
}
// order-preserving float <-> uint encoding for atomicMax on floats
__device__ __forceinline__ unsigned fenc(float f) {
  unsigned u = __float_as_uint(f);
  return (u & 0x80000000u) ? ~u : (u | 0x80000000u);
}
__device__ __forceinline__ float fdec(unsigned u) {
  unsigned v = (u & 0x80000000u) ? (u & 0x7fffffffu) : ~u;
  return __uint_as_float(v);
}
#define ENC_NEG_INF 0x007fffffu   // fenc(-inf)

__device__ __forceinline__ void atomic_add_f32(float* p, float v) {
  unsafeAtomicAdd(p, v);          // hw global_atomic_add_f32
}

// one Chen step: order matters (s3 uses old s2/s1; s2 uses old s1)
__device__ __forceinline__ void sig_step(float* s1, float* s2, float* s3,
                                         float d0, float d1, float d2, float d3) {
  float dx[4] = {d0, d1, d2, d3};
  float dd[16];
#pragma unroll
  for (int a = 0; a < 4; ++a)
#pragma unroll
    for (int b = 0; b < 4; ++b) dd[a * 4 + b] = dx[a] * dx[b];
#pragma unroll
  for (int a = 0; a < 4; ++a)
#pragma unroll
    for (int b = 0; b < 4; ++b)
#pragma unroll
      for (int c = 0; c < 4; ++c)
        s3[a * 16 + b * 4 + c] += s2[a * 4 + b] * dx[c]
                                + 0.5f * s1[a] * dd[b * 4 + c]
                                + (1.0f / 6.0f) * dx[a] * dd[b * 4 + c];
#pragma unroll
  for (int a = 0; a < 4; ++a)
#pragma unroll
    for (int b = 0; b < 4; ++b)
      s2[a * 4 + b] += s1[a] * dx[b] + 0.5f * dd[a * 4 + b];
#pragma unroll
  for (int a = 0; a < 4; ++a) s1[a] += dx[a];
}

// ---------------- kernels ----------------

// signatures (3 windows x 84) + categorical embeddings -> bf16 input rows (padded)
__global__ void build_input_kernel(const float* __restrict__ xf,
                                   const int* __restrict__ store_id,
                                   const int* __restrict__ dept_id,
                                   const int* __restrict__ cat_id,
                                   const int* __restrict__ state_id,
                                   const int* __restrict__ item_id,
                                   const float* __restrict__ emb_store,
                                   const float* __restrict__ emb_dept,
                                   const float* __restrict__ emb_cat,
                                   const float* __restrict__ emb_state,
                                   const float* __restrict__ emb_item,
                                   __bf16* __restrict__ Xin) {
  int n = blockIdx.x * blockDim.x + threadIdx.x;
  if (n >= NPAD) return;
  __bf16* row = Xin + (size_t)n * KIN;
  if (n >= NREAL) {
    for (int j = 0; j < KIN; ++j) row[j] = (__bf16)0.0f;
    return;
  }
  const float* x = xf + (size_t)n * TDIM * CIN;
  const int wins[3] = {7, 28, 90};
  int off = 0;
  for (int wi = 0; wi < 3; ++wi) {
    int w = wins[wi], t0 = TDIM - w;
    float s1[4], s2[16], s3[64];
#pragma unroll
    for (int j = 0; j < 4; ++j) s1[j] = 0.f;
#pragma unroll
    for (int j = 0; j < 16; ++j) s2[j] = 0.f;
#pragma unroll
    for (int j = 0; j < 64; ++j) s3[j] = 0.f;
    for (int m = 0; m < w - 1; ++m) {
      float d0 = x[(t0 + m + 1) * CIN + 0] - x[(t0 + m) * CIN + 0];
      float d1 = x[(t0 + m + 1) * CIN + 1] - x[(t0 + m) * CIN + 1];
      sig_step(s1, s2, s3, d0, d1, 0.f, 0.f);   // lead-lag even step
      sig_step(s1, s2, s3, 0.f, 0.f, d0, d1);   // lead-lag odd step
    }
#pragma unroll
    for (int j = 0; j < 4; ++j)  row[off + j]      = (__bf16)s1[j];
#pragma unroll
    for (int j = 0; j < 16; ++j) row[off + 4 + j]  = (__bf16)s2[j];
#pragma unroll
    for (int j = 0; j < 64; ++j) row[off + 20 + j] = (__bf16)s3[j];
    off += 84;
  }
  const float* e0 = emb_store + store_id[n] * 8;
  const float* e1 = emb_dept  + dept_id[n]  * 8;
  const float* e2 = emb_cat   + cat_id[n]   * 4;
  const float* e3 = emb_state + state_id[n] * 4;
  const float* e4 = emb_item  + item_id[n]  * 16;
#pragma unroll
  for (int j = 0; j < 8; ++j)  row[off + j]      = (__bf16)e0[j];
#pragma unroll
  for (int j = 0; j < 8; ++j)  row[off + 8 + j]  = (__bf16)e1[j];
#pragma unroll
  for (int j = 0; j < 4; ++j)  row[off + 16 + j] = (__bf16)e2[j];
#pragma unroll
  for (int j = 0; j < 4; ++j)  row[off + 20 + j] = (__bf16)e3[j];
#pragma unroll
  for (int j = 0; j < 16; ++j) row[off + 24 + j] = (__bf16)e4[j];
  for (int j = off + 40; j < KIN; ++j) row[j] = (__bf16)0.0f;   // pad 292..319
}

// pack f32 weights (Ks x Ns, row-major) into WMMA-B fragment order, bf16, padded
// to Kp x Np.  Layout: for k-tile kt (32 rows) and n-tile nt (16 cols), the 16
// bf16 values lane L needs are stored contiguously:
//   Bp[(((kt*(Np/16))+nt)*32 + L)*16 + e] = B[kt*32 + ksel(L,e), nt*16 + (L&15)]
// with ksel(L,e) = (L>>4)*8 + (e<8 ? e : e+8)   (ISA 16-bit B layout)
__global__ void pack_weight_kernel(const float* __restrict__ src, __bf16* __restrict__ dst,
                                   int Ks, int Ns, int Kp, int Np) {
  int i = blockIdx.x * blockDim.x + threadIdx.x;
  if (i >= Kp * Np) return;
  int e    = i & 15;
  int lane = (i >> 4) & 31;
  int tile = i >> 9;
  int ntiles = Np >> 4;
  int kt = tile / ntiles, nt = tile - kt * ntiles;
  int k = kt * 32 + (lane >> 4) * 8 + (e < 8 ? e : e + 8);
  int n = nt * 16 + (lane & 15);
  dst[i] = (k < Ks && n < Ns) ? (__bf16)src[k * Ns + n] : (__bf16)0.0f;
}

// WMMA bf16 GEMM: C[NPAD x Nc] = act(A[NPAD x K] * B + bias + extra)
// one wave computes a 16x32 tile (two 16x16 WMMA tiles sharing the A fragment).
// A: row-major bf16; Bp: packed fragment layout (see pack_weight_kernel).
// act: 0=none, 1=gelu, 2=elu
__global__ __launch_bounds__(32)
void gemm_wmma_kernel(const __bf16* __restrict__ A, const __bf16* __restrict__ Bp,
                      const float* __restrict__ bias, const float* __restrict__ extra,
                      float* __restrict__ C, int K, int Nc, int act) {
  int m0  = blockIdx.x * 16;
  int nt0 = blockIdx.y * 2;             // two consecutive 16-col tiles
  int lane = threadIdx.x;
  int half = lane >> 4;
  int l15  = lane & 15;
  int ntiles = Nc >> 4;
  v8f acc0 = {}, acc1 = {};
  const __bf16* arow = A + (size_t)(m0 + l15) * K + half * 8;
  const __bf16* bbase = Bp + ((size_t)nt0 * 32 + lane) * 16;
  size_t bstride = (size_t)ntiles * 32 * 16;   // elements per k-tile of packed B
  for (int k0 = 0; k0 < K; k0 += 32) {
    Frag16 fa, fb0, fb1;
    fa.q[0] = *(const uint4*)(arow + k0);
    fa.q[1] = *(const uint4*)(arow + k0 + 16);
    const __bf16* bp = bbase + (size_t)(k0 >> 5) * bstride;
    fb0.q[0] = *(const uint4*)(bp);
    fb0.q[1] = *(const uint4*)(bp + 8);
    fb1.q[0] = *(const uint4*)(bp + 512);        // next 16-col tile: +32 lanes*16
    fb1.q[1] = *(const uint4*)(bp + 520);
    acc0 = __builtin_amdgcn_wmma_f32_16x16x32_bf16(false, fa.v, false, fb0.v,
                                                   (short)0, acc0, false, false);
    acc1 = __builtin_amdgcn_wmma_f32_16x16x32_bf16(false, fa.v, false, fb1.v,
                                                   (short)0, acc1, false, false);
  }
  int mbase = m0 + half * 8;
#pragma unroll
  for (int t = 0; t < 2; ++t) {
    int col = (nt0 + t) * 16 + l15;
    float bv = bias ? bias[col] : 0.0f;
    v8f acc = t ? acc1 : acc0;
#pragma unroll
    for (int r = 0; r < 8; ++r) {
      int row = mbase + r;
      float v = acc[r] + bv;
      if (extra) v += extra[(size_t)row * Nc + col];
      if (act == 1)      v = gelu_f(v);
      else if (act == 2) v = (v > 0.f) ? v : expm1f(v);   // ELU
      C[(size_t)row * Nc + col] = v;
    }
  }
}

__global__ void layernorm_kernel(const float* __restrict__ in, const float* __restrict__ g,
                                 const float* __restrict__ b, __bf16* __restrict__ hb) {
  int n = blockIdx.x * blockDim.x + threadIdx.x;
  if (n >= NPAD) return;
  const float* r = in + (size_t)n * HID;
  float mean = 0.f;
  for (int j = 0; j < HID; ++j) mean += r[j];
  mean *= (1.0f / HID);
  float var = 0.f;
  for (int j = 0; j < HID; ++j) { float d = r[j] - mean; var += d * d; }
  var *= (1.0f / HID);
  float inv = rsqrtf(var + 1e-5f);
  for (int j = 0; j < HID; ++j)
    hb[(size_t)n * HID + j] = (__bf16)((r[j] - mean) * inv * g[j] + b[j]);
}

__global__ void attn_coef_kernel(const float* __restrict__ q, const float* __restrict__ asrc,
                                 const float* __restrict__ adst, float* __restrict__ es,
                                 float* __restrict__ ed) {
  int idx = blockIdx.x * blockDim.x + threadIdx.x;
  if (idx >= NPAD * NH) return;
  int n = idx >> 2, h = idx & 3;
  const float* qp = q + (size_t)n * HID + h * DHD;
  float ss = 0.f, sd = 0.f;
#pragma unroll
  for (int d = 0; d < DHD; ++d) { ss += qp[d] * asrc[h * DHD + d]; sd += qp[d] * adst[h * DHD + d]; }
  es[idx] = ss; ed[idx] = sd;
}

__global__ void init_seg_kernel(unsigned* __restrict__ segmax, float* __restrict__ den) {
  int i = blockIdx.x * blockDim.x + threadIdx.x;
  if (i >= NPAD * NH) return;
  segmax[i] = ENC_NEG_INF;
  den[i] = 0.0f;
}

__global__ void zero_f32_kernel(float* __restrict__ p, int count) {
  int i = blockIdx.x * blockDim.x + threadIdx.x;
  if (i < count) p[i] = 0.0f;
}

__global__ void edge_max_kernel(const int* __restrict__ src, const int* __restrict__ dst,
                                const int* __restrict__ et, const float* __restrict__ es,
                                const float* __restrict__ ed, const float* __restrict__ etb,
                                unsigned* __restrict__ segmax) {
  int e = blockIdx.x * blockDim.x + threadIdx.x;
  if (e >= EEDGE) return;
  int s = src[e], d = dst[e], t = et[e];
#pragma unroll
  for (int h = 0; h < NH; ++h) {
    float l = es[s * NH + h] + ed[d * NH + h] + etb[t * NH + h];
    l = (l > 0.f) ? l : 0.2f * l;                     // leaky_relu(0.2)
    atomicMax(&segmax[d * NH + h], fenc(l));
  }
}

__global__ void seg_decode_kernel(const unsigned* __restrict__ segmax, float* __restrict__ m) {
  int i = blockIdx.x * blockDim.x + threadIdx.x;
  if (i >= NPAD * NH) return;
  float v = fdec(segmax[i]);
  m[i] = isfinite(v) ? v : 0.0f;
}

__global__ void edge_exp_kernel(const int* __restrict__ src, const int* __restrict__ dst,
                                const int* __restrict__ et, const float* __restrict__ es,
                                const float* __restrict__ ed, const float* __restrict__ etb,
                                const float* __restrict__ m, float* __restrict__ exbuf,
                                float* __restrict__ den) {
  int e = blockIdx.x * blockDim.x + threadIdx.x;
  if (e >= EEDGE) return;
  int s = src[e], d = dst[e], t = et[e];
#pragma unroll
  for (int h = 0; h < NH; ++h) {
    float l = es[s * NH + h] + ed[d * NH + h] + etb[t * NH + h];
    l = (l > 0.f) ? l : 0.2f * l;
    float ex = __expf(l - m[d * NH + h]);
    exbuf[e * NH + h] = ex;
    atomic_add_f32(&den[d * NH + h], ex);
  }
}

__global__ void edge_msg_kernel(const int* __restrict__ src, const int* __restrict__ dst,
                                const float* __restrict__ exbuf, const float* __restrict__ den,
                                const float* __restrict__ q, float* __restrict__ msg) {
  int i = blockIdx.x * blockDim.x + threadIdx.x;
  if (i >= EEDGE * HID) return;
  int e = i >> 7, k = i & 127, h = k >> 5;
  int d = dst[e];
  float a = exbuf[e * NH + h] / (den[d * NH + h] + 1e-9f);
  atomic_add_f32(&msg[(size_t)d * HID + k], a * q[(size_t)src[e] * HID + k]);
}

__global__ void f32_to_bf16_kernel(const float* __restrict__ src, __bf16* __restrict__ dst,
                                   int count) {
  int i = blockIdx.x * blockDim.x + threadIdx.x;
  if (i < count) dst[i] = (__bf16)src[i];
}

__global__ void pad_bias_kernel(const float* __restrict__ src, float* __restrict__ dst,
                                int n, int np) {
  int i = blockIdx.x * blockDim.x + threadIdx.x;
  if (i < np) dst[i] = (i < n) ? src[i] : 0.0f;
}

__global__ void finalize_kernel(const float* __restrict__ x3, const float* __restrict__ hscale,
                                const float* __restrict__ gscale, const int* __restrict__ dept_ids,
                                const float* __restrict__ hist, float* __restrict__ out) {
  int idx = blockIdx.x * blockDim.x + threadIdx.x;
  if (idx >= NREAL * HOR) return;
  int n = idx / HOR, j = idx - n * HOR;
  float sp1 = log1pf(__expf(hscale[j]));                 // softplus
  float sp2 = log1pf(__expf(gscale[dept_ids[n]]));
  float v = x3[(size_t)n * HORP + j] * sp1 * sp2;
  float hi = 20.0f * hist[n];
  v = fminf(fmaxf(v, 0.0f), hi);
  out[idx] = v;
}

// ---------------- host launcher ----------------
extern "C" void kernel_launch(void* const* d_in, const int* in_sizes, int n_in,
                              void* d_out, int out_size, void* d_ws, size_t ws_size,
                              hipStream_t stream) {
  (void)in_sizes; (void)n_in; (void)out_size; (void)ws_size;
  const float* node_features  = (const float*)d_in[0];
  const int*   edge_index     = (const int*)d_in[1];
  const int*   edge_type      = (const int*)d_in[2];
  const int*   store_id       = (const int*)d_in[3];
  const int*   dept_id        = (const int*)d_in[4];
  const int*   cat_id         = (const int*)d_in[5];
  const int*   state_id       = (const int*)d_in[6];
  const int*   item_id        = (const int*)d_in[7];
  const int*   dept_ids       = (const int*)d_in[8];
  const float* historical_mean= (const float*)d_in[9];
  const float* emb_store      = (const float*)d_in[10];
  const float* emb_dept       = (const float*)d_in[11];
  const float* emb_cat        = (const float*)d_in[12];
  const float* emb_state      = (const float*)d_in[13];
  const float* emb_item       = (const float*)d_in[14];
  const float* fusion_W       = (const float*)d_in[15];
  const float* fusion_b       = (const float*)d_in[16];
  const float* ln_g           = (const float*)d_in[17];
  const float* ln_b           = (const float*)d_in[18];
  const float* gat_W          = (const float*)d_in[19];
  const float* gat_b          = (const float*)d_in[20];
  const float* gat_asrc       = (const float*)d_in[21];
  const float* gat_adst       = (const float*)d_in[22];
  const float* gat_et         = (const float*)d_in[23];
  const float* gat_Wres       = (const float*)d_in[24];
  const float* pred_W1        = (const float*)d_in[25];
  const float* pred_b1        = (const float*)d_in[26];
  const float* pred_W2        = (const float*)d_in[27];
  const float* pred_b2        = (const float*)d_in[28];
  const float* pred_W3        = (const float*)d_in[29];
  const float* pred_b3        = (const float*)d_in[30];
  const float* horizon_scale  = (const float*)d_in[31];
  const float* group_scale    = (const float*)d_in[32];

  const int* e_src = edge_index;
  const int* e_dst = edge_index + EEDGE;

  // bump allocator over d_ws
  char* base = (char*)d_ws;
  size_t cur = 0;
  auto alloc = [&](size_t bytes) -> void* {
    void* p = base + cur;
    cur += (bytes + 255) & ~(size_t)255;
    return p;
  };
  __bf16* Xin      = (__bf16*)alloc((size_t)NPAD * KIN * 2);
  __bf16* Wfus     = (__bf16*)alloc((size_t)KIN * HID * 2);
  __bf16* gatWb    = (__bf16*)alloc((size_t)LAYERS * HID * HID * 2);
  __bf16* gatWresb = (__bf16*)alloc((size_t)LAYERS * HID * HID * 2);
  __bf16* W1b      = (__bf16*)alloc((size_t)HID * MLPH * 2);
  __bf16* W2b      = (__bf16*)alloc((size_t)MLPH * MLPH * 2);
  __bf16* W3b      = (__bf16*)alloc((size_t)MLPH * HORP * 2);
  float*  b3p      = (float*) alloc((size_t)HORP * 4);
  float*  hpre     = (float*) alloc((size_t)NPAD * HID * 4);
  __bf16* hb       = (__bf16*)alloc((size_t)NPAD * HID * 2);
  float*  qf       = (float*) alloc((size_t)NPAD * HID * 4);
  float*  msg      = (float*) alloc((size_t)NPAD * HID * 4);
  float*  esb      = (float*) alloc((size_t)NPAD * NH * 4);
  float*  edb      = (float*) alloc((size_t)NPAD * NH * 4);
  unsigned* segmax = (unsigned*)alloc((size_t)NPAD * NH * 4);
  float*  mbuf     = (float*) alloc((size_t)NPAD * NH * 4);
  float*  den      = (float*) alloc((size_t)NPAD * NH * 4);
  float*  exbuf    = (float*) alloc((size_t)EEDGE * NH * 4);
  float*  x1f      = (float*) alloc((size_t)NPAD * MLPH * 4);
  __bf16* x1b      = (__bf16*)alloc((size_t)NPAD * MLPH * 2);
  float*  x2f      = (float*) alloc((size_t)NPAD * MLPH * 4);
  __bf16* x2b      = (__bf16*)alloc((size_t)NPAD * MLPH * 2);
  float*  x3f      = (float*) alloc((size_t)NPAD * HORP * 4);

  auto blocks = [](long n) { return (unsigned)((n + 255) / 256); };
  auto gemm = [&](const __bf16* A, const __bf16* Bp, const float* bias, const float* extra,
                  float* Cc, int K, int Nc, int act) {
    dim3 grid(NPAD / 16, Nc / 32), blk(32);
    gemm_wmma_kernel<<<grid, blk, 0, stream>>>(A, Bp, bias, extra, Cc, K, Nc, act);
  };
  auto pack = [&](const float* src, __bf16* dst, int Ks, int Ns, int Kp, int Np) {
    pack_weight_kernel<<<blocks((long)Kp * Np), 256, 0, stream>>>(src, dst, Ks, Ns, Kp, Np);
  };

  // ---- weight packing (f32 -> fragment-ordered bf16) ----
  pack(fusion_W, Wfus, 292, HID, KIN, HID);
  for (int i = 0; i < LAYERS; ++i) {
    pack(gat_W + (size_t)i * HID * HID, gatWb + (size_t)i * HID * HID, HID, HID, HID, HID);
    pack(gat_Wres + (size_t)i * HID * HID, gatWresb + (size_t)i * HID * HID, HID, HID, HID, HID);
  }
  pack(pred_W1, W1b, HID, MLPH, HID, MLPH);
  pack(pred_W2, W2b, MLPH, MLPH, MLPH, MLPH);
  pack(pred_W3, W3b, MLPH, HOR, MLPH, HORP);
  pad_bias_kernel<<<1, 64, 0, stream>>>(pred_b3, b3p, HOR, HORP);

  // ---- features: signatures + embeddings ----
  build_input_kernel<<<blocks(NPAD), 256, 0, stream>>>(
      node_features, store_id, dept_id, cat_id, state_id, item_id,
      emb_store, emb_dept, emb_cat, emb_state, emb_item, Xin);

  // ---- fusion: gelu(X @ Wf + bf) -> layernorm -> hb ----
  gemm(Xin, Wfus, fusion_b, nullptr, hpre, KIN, HID, 1);
  layernorm_kernel<<<blocks(NPAD), 256, 0, stream>>>(hpre, ln_g, ln_b, hb);

  // ---- GAT layers ----
  for (int i = 0; i < LAYERS; ++i) {
    const __bf16* Wi  = gatWb    + (size_t)i * HID * HID;
    const __bf16* Wri = gatWresb + (size_t)i * HID * HID;
    const float*  bi  = gat_b    + (size_t)i * HID;
    const float*  asi = gat_asrc + (size_t)i * NH * DHD;
    const float*  adi = gat_adst + (size_t)i * NH * DHD;
    const float*  eti = gat_et   + (size_t)i * ETYPES * NH;

    gemm(hb, Wi, bi, nullptr, qf, HID, HID, 0);                       // q = h@W + b
    attn_coef_kernel<<<blocks((long)NPAD * NH), 256, 0, stream>>>(qf, asi, adi, esb, edb);
    init_seg_kernel<<<blocks((long)NPAD * NH), 256, 0, stream>>>(segmax, den);
    zero_f32_kernel<<<blocks((long)NPAD * HID), 256, 0, stream>>>(msg, NPAD * HID);
    edge_max_kernel<<<blocks(EEDGE), 256, 0, stream>>>(e_src, e_dst, edge_type, esb, edb, eti, segmax);
    seg_decode_kernel<<<blocks((long)NPAD * NH), 256, 0, stream>>>(segmax, mbuf);
    edge_exp_kernel<<<blocks(EEDGE), 256, 0, stream>>>(e_src, e_dst, edge_type, esb, edb, eti,
                                                       mbuf, exbuf, den);
    edge_msg_kernel<<<blocks((long)EEDGE * HID), 256, 0, stream>>>(e_src, e_dst, exbuf, den, qf, msg);
    gemm(hb, Wri, nullptr, msg, hpre, HID, HID, 2);                   // elu(msg + h@Wres)
    f32_to_bf16_kernel<<<blocks((long)NPAD * HID), 256, 0, stream>>>(hpre, hb, NPAD * HID);
  }

  // ---- prediction MLP ----
  gemm(hb, W1b, pred_b1, nullptr, x1f, HID, MLPH, 1);
  f32_to_bf16_kernel<<<blocks((long)NPAD * MLPH), 256, 0, stream>>>(x1f, x1b, NPAD * MLPH);
  gemm(x1b, W2b, pred_b2, nullptr, x2f, MLPH, MLPH, 1);
  f32_to_bf16_kernel<<<blocks((long)NPAD * MLPH), 256, 0, stream>>>(x2f, x2b, NPAD * MLPH);
  gemm(x2b, W3b, b3p, nullptr, x3f, MLPH, HORP, 0);

  finalize_kernel<<<blocks((long)NREAL * HOR), 256, 0, stream>>>(
      x3f, horizon_scale, group_scale, dept_ids, historical_mean, (float*)d_out);
}
